// GCNNet_26474178413326
// MI455X (gfx1250) — matrix operand order repaired and checked by reading, not compiled, not run on hardware
//
#include <hip/hip_runtime.h>
#include <hip/hip_bf16.h>

// ---------------------------------------------------------------------------
// GCNNet on MI455X (gfx1250, wave32).
//  - Big GEMM h = x @ W1 uses v_wmma_f32_16x16x32_f16 (f16 in, f32 acc).
//    Fragments assembled with 16B ds_load_b128 pairs (A from row-major sX,
//    B from transposed sWt). D stores: one wave-uniform bound branch around
//    unguarded coalesced stores (N_NODES % 16 == 0 -> no partial waves).
//  - Edge scatter / degree use hardware f32 global atomics (L2-resident:
//    h+agg+dinv ~26MB << 192MB L2; nets processed sequentially reusing one
//    scratch pair keeps the 50M atomics/net off HBM).
//  - Segment-max uses uint atomicMax on non-negative floats (post-ReLU),
//    init 0 == reference's where(isneginf, 0).
// ---------------------------------------------------------------------------

typedef __attribute__((ext_vector_type(16))) _Float16 v16h;
typedef __attribute__((ext_vector_type(8)))  _Float16 v8h;
typedef __attribute__((ext_vector_type(8)))  float    v8f;

#define N_NODES 50000
#define N_EDGES 800000
#define N_GRAPHS 512
#define F_IN 62
#define F_OUT 64
#define FP 64          // padded feature stride
#define KS 72          // LDS K-stride in f16 (144B: 16B-aligned chunks)

// Two aligned 16B LDS loads -> one v16h WMMA operand.
__device__ inline v16h ld_frag(const _Float16* p0, const _Float16* p1) {
  v8h lo = *(const v8h*)p0;
  v8h hi = *(const v8h*)p1;
  return __builtin_shufflevector(lo, hi, 0,1,2,3,4,5,6,7,8,9,10,11,12,13,14,15);
}

// ---------------- utility ----------------
__global__ void zero_f32_kernel(float* __restrict__ p, int n) {
  int i = blockIdx.x * blockDim.x + threadIdx.x;
  if (i < n) p[i] = 0.0f;
}

// ---------------- degree / dinv ----------------
__global__ void deg_kernel(const int* __restrict__ col, float* __restrict__ deg, int n_edges) {
  int e = blockIdx.x * blockDim.x + threadIdx.x;
  if (e < n_edges) {
    __hip_atomic_fetch_add(&deg[col[e]], 1.0f, __ATOMIC_RELAXED, __HIP_MEMORY_SCOPE_AGENT);
  }
}

__global__ void rsqrt_kernel(float* __restrict__ d, int n) {
  int i = blockIdx.x * blockDim.x + threadIdx.x;
  if (i < n) d[i] = rsqrtf(d[i] + 1.0f);   // +1 self loop
}

// ---------------- WMMA GEMM: h[n,0:64] = x[n,0:62] @ W[62x62 zero-padded] ----
// Block: 256 threads = 8 waves; each wave computes a 16x64 strip.
// REQUIRES n_rows % 16 == 0 (holds: N_NODES = 50000 = 3125*16).
__global__ __launch_bounds__(256) void wmma_xw_kernel(
    const float* __restrict__ x,   // [N_NODES, 62]
    const float* __restrict__ W,   // [62, 62] row-major (K x N)
    float* __restrict__ h,         // [N_NODES, 64]
    int n_rows)
{
  __shared__ _Float16 sX[128][KS];   // [row][K]  (row-major, K padded 62->64)
  __shared__ _Float16 sWt[64][KS];   // [N][K]    (W transposed)

  const int tid = threadIdx.x;
  const int block_m = blockIdx.x * 128;

  // Stage W transposed (62x62 -> [N=64][K=64], zero padded), f32 -> f16
  for (int idx = tid; idx < 64 * 64; idx += 256) {
    int k = idx >> 6, n = idx & 63;
    float v = (k < F_IN && n < F_IN) ? W[k * F_IN + n] : 0.0f;
    sWt[n][k] = (_Float16)v;
  }
  // Stage x tile (128x62 -> 128x64)
  for (int idx = tid; idx < 128 * 64; idx += 256) {
    int r = idx >> 6, c = idx & 63;
    int gr = block_m + r;
    float v = (gr < n_rows && c < F_IN) ? x[gr * F_IN + c] : 0.0f;
    sX[r][c] = (_Float16)v;
  }
  __syncthreads();

  const int wave  = tid >> 5;
  const int lane  = tid & 31;
  const int l15   = lane & 15;   // A row M, B col N, D col N
  const int khalf = lane >> 4;

  v8f acc[4] = {v8f{}, v8f{}, v8f{}, v8f{}};

  #pragma unroll
  for (int kk = 0; kk < 64; kk += 32) {
    // A fragment (16-bit 16x32 layout): per lane two contiguous 8-K chunks:
    //   j=0..7  -> K = kk + khalf*8 + j
    //   j=8..15 -> K = kk + 16 + khalf*8 + j
    const _Float16* arow = &sX[wave * 16 + l15][0];
    v16h a = ld_frag(arow + kk + khalf * 8, arow + kk + 16 + khalf * 8);

    #pragma unroll
    for (int t = 0; t < 4; ++t) {
      // B fragment (32x16): N = t*16 + l15, K same as A (two 8-K chunks at khalf*8 and 16 + khalf*8)
      const _Float16* brow = &sWt[t * 16 + l15][0];
      v16h b = ld_frag(brow + kk + khalf * 8, brow + kk + 16 + khalf * 8);
      acc[t] = __builtin_amdgcn_wmma_f32_16x16x32_f16(
          false, a, false, b, (short)0, acc[t], false, false);
    }
  }

  // D layout: VGPR v -> M = khalf*8 + v, N = l15 (+16t).
  // n_rows % 16 == 0: a wave's 16 rows are all-valid or all-invalid, so a
  // single wave-uniform scalar branch suffices; stores are unguarded and
  // coalesced (16 lanes -> 64 contiguous bytes per row).
  const int wave_m0 = block_m + wave * 16;
  if (__builtin_amdgcn_readfirstlane((wave_m0 + 16 <= n_rows) ? 1 : 0)) {
    #pragma unroll
    for (int t = 0; t < 4; ++t) {
      #pragma unroll
      for (int v = 0; v < 8; ++v) {
        h[(wave_m0 + khalf * 8 + v) * FP + t * 16 + l15] = acc[t][v];
      }
    }
  }
}

// ---------------- edge scatter: agg[col] += h[row] * dinv[row]*dinv[col] ----
__global__ void scatter_kernel(const int* __restrict__ row, const int* __restrict__ col,
                               const float* __restrict__ h, const float* __restrict__ dinv,
                               float* __restrict__ agg, int n_edges)
{
  int idx = blockIdx.x * blockDim.x + threadIdx.x;   // (edge, feature)
  int e = idx >> 6, f = idx & 63;
  if (e >= n_edges || f >= F_IN) return;
  int r = row[e], c = col[e];
  float nrm = dinv[r] * dinv[c];
  __hip_atomic_fetch_add(&agg[c * FP + f], h[r * FP + f] * nrm,
                         __ATOMIC_RELAXED, __HIP_MEMORY_SCOPE_AGENT);
}

// ---------------- self loop + bias + relu (in place into h) ----------------
__global__ void combine_kernel(const float* __restrict__ agg, float* __restrict__ h,
                               const float* __restrict__ dinv, const float* __restrict__ b,
                               int n_nodes)
{
  int idx = blockIdx.x * blockDim.x + threadIdx.x;
  int n = idx >> 6, f = idx & 63;
  if (n >= n_nodes) return;
  float v = 0.0f;
  if (f < F_IN) {
    float di = dinv[n];
    v = fmaxf(agg[idx] + h[idx] * di * di + b[f], 0.0f);
  }
  h[idx] = v;
}

// ---------------- segment max (global max pool) -----------------------------
// h >= 0 post-relu -> uint bit-pattern compare == float compare; init 0.
__global__ void segmax_kernel(const float* __restrict__ h, const int* __restrict__ batch,
                              unsigned* __restrict__ gmax, int n_nodes)
{
  int idx = blockIdx.x * blockDim.x + threadIdx.x;
  int n = idx >> 6, f = idx & 63;
  if (n >= n_nodes || f >= F_IN) return;
  unsigned v = __float_as_uint(h[idx]);
  __hip_atomic_fetch_max(&gmax[batch[n] * FP + f], v,
                         __ATOMIC_RELAXED, __HIP_MEMORY_SCOPE_AGENT);
}

// ---------------- g = relu(gmax @ Wg + bg); acc (+)= 0.25*g ------------------
__global__ void graph_mlp_kernel(const unsigned* __restrict__ gmax,
                                 const float* __restrict__ Wg, const float* __restrict__ bg,
                                 float* __restrict__ acc, int first)
{
  int idx = blockIdx.x * blockDim.x + threadIdx.x;   // G*64
  int g = idx >> 6, f = idx & 63;
  if (g >= N_GRAPHS) return;
  float s = bg[f];
  #pragma unroll 2
  for (int k = 0; k < F_IN; ++k)
    s += __uint_as_float(gmax[g * FP + k]) * Wg[k * F_OUT + f];
  s = fmaxf(s, 0.0f) * 0.25f;
  if (first) acc[idx] = s; else acc[idx] += s;
}

// ---------------- xc = relu(acc @ fc1_W + fc1_b) -----------------------------
__global__ void fc1_kernel(const float* __restrict__ acc, const float* __restrict__ W,
                           const float* __restrict__ b, float* __restrict__ xc)
{
  int idx = blockIdx.x * blockDim.x + threadIdx.x;
  int g = idx >> 6, f = idx & 63;
  if (g >= N_GRAPHS) return;
  float s = b[f];
  #pragma unroll 4
  for (int k = 0; k < 64; ++k) s += acc[g * 64 + k] * W[k * 64 + f];
  xc[idx] = fmaxf(s, 0.0f);
}

// ---------------- out = xc @ out_W + out_b  [G,1] ----------------------------
__global__ void out_kernel(const float* __restrict__ xc, const float* __restrict__ W,
                           const float* __restrict__ b, float* __restrict__ out)
{
  int g = blockIdx.x * blockDim.x + threadIdx.x;
  if (g >= N_GRAPHS) return;
  float s = b[0];
  #pragma unroll 4
  for (int k = 0; k < 64; ++k) s += xc[g * 64 + k] * W[k];
  out[g] = s;
}

// ---------------------------------------------------------------------------
extern "C" void kernel_launch(void* const* d_in, const int* in_sizes, int n_in,
                              void* d_out, int out_size, void* d_ws, size_t ws_size,
                              hipStream_t stream)
{
  const int N = N_NODES, E = N_EDGES, G = N_GRAPHS;

  // Workspace carve (~26.3 MB), reused across the 4 nets (stays L2-resident).
  char* ws = (char*)d_ws;
  float*    h    = (float*)ws;     ws += (size_t)N * FP * sizeof(float);
  float*    agg  = (float*)ws;     ws += (size_t)N * FP * sizeof(float);
  float*    dinv = (float*)ws;     ws += (size_t)((N + 63) & ~63) * sizeof(float);
  unsigned* gmax = (unsigned*)ws;  ws += (size_t)G * FP * sizeof(float);
  float*    acc  = (float*)ws;     ws += (size_t)G * FP * sizeof(float);
  float*    xc   = (float*)ws;     ws += (size_t)G * FP * sizeof(float);

  // setup_inputs() dict order: (x_i, edge_index_i, batch_i) x4, then params.
  const float* W1   = (const float*)d_in[12];  // [4,62,62]
  const float* b1   = (const float*)d_in[13];  // [4,62]
  const float* Wg   = (const float*)d_in[14];  // [4,62,64]
  const float* bg   = (const float*)d_in[15];  // [4,64]
  const float* fc1W = (const float*)d_in[16];  // [64,64]
  const float* fc1b = (const float*)d_in[17];  // [64]
  const float* outW = (const float*)d_in[18];  // [64,1]
  const float* outb = (const float*)d_in[19];  // [1]

  const int T = 256;
  for (int i = 0; i < 4; ++i) {
    const float* x     = (const float*)d_in[3 * i + 0];
    const int*   ei    = (const int*)  d_in[3 * i + 1];
    const int*   batch = (const int*)  d_in[3 * i + 2];
    const int* row = ei;        // edge_index[0]
    const int* col = ei + E;    // edge_index[1]

    zero_f32_kernel<<<(N * FP + T - 1) / T, T, 0, stream>>>(agg, N * FP);
    zero_f32_kernel<<<(N + T - 1) / T, T, 0, stream>>>(dinv, N);          // deg accumulator
    zero_f32_kernel<<<(G * FP + T - 1) / T, T, 0, stream>>>((float*)gmax, G * FP);

    deg_kernel<<<(E + T - 1) / T, T, 0, stream>>>(col, dinv, E);
    rsqrt_kernel<<<(N + T - 1) / T, T, 0, stream>>>(dinv, N);

    wmma_xw_kernel<<<(N + 127) / 128, 256, 0, stream>>>(x, W1 + (size_t)i * F_IN * F_IN, h, N);

    scatter_kernel<<<((E * 64) + T - 1) / T, T, 0, stream>>>(row, col, h, dinv, agg, E);
    combine_kernel<<<(N * FP + T - 1) / T, T, 0, stream>>>(agg, h, dinv, b1 + (size_t)i * F_IN, N);
    segmax_kernel<<<(N * FP + T - 1) / T, T, 0, stream>>>(h, batch, gmax, N);
    graph_mlp_kernel<<<(G * FP + T - 1) / T, T, 0, stream>>>(
        gmax, Wg + (size_t)i * F_IN * F_OUT, bg + (size_t)i * F_OUT, acc, (i == 0) ? 1 : 0);
  }

  fc1_kernel<<<(G * 64 + T - 1) / T, T, 0, stream>>>(acc, fc1W, fc1b, xc);
  out_kernel<<<(G + T - 1) / T, T, 0, stream>>>(xc, outW, outb, (float*)d_out);
}
